// NonTransition_56538949485053
// MI455X (gfx1250) — matrix operand order repaired and checked by reading, not compiled, not run on hardware
//
#include <hip/hip_runtime.h>
#include <hip/hip_bf16.h>

#define BATCH 8
#define NPTS  2048
#define CIN   64
#define COUT  128
#define KNN   16
#define EPS   1e-5f

typedef _Float16 v16h __attribute__((ext_vector_type(16)));
typedef _Float16 v8h  __attribute__((ext_vector_type(8)));
typedef float    v8f  __attribute__((ext_vector_type(8)));
typedef float    v4f  __attribute__((ext_vector_type(4)));

// ---------------------------------------------------------------------------
// Kernel 1: KNN (top-16 by squared distance, self included) + local_coords.
// One thread per query point; candidate coords tiled through LDS.
// ---------------------------------------------------------------------------
__global__ __launch_bounds__(256) void knn_local_kernel(
    const float* __restrict__ coords,   // [B, 3, N]
    int*         __restrict__ idx_out,  // [B, N, K]   (workspace)
    float*       __restrict__ lc_out)   // [B, 3, N, K] (output section)
{
    __shared__ float cs0[256], cs1[256], cs2[256];

    const int tid = threadIdx.x;
    const int b   = blockIdx.x >> 3;                  // 8 blocks of 256 pts per batch
    const int n   = ((blockIdx.x & 7) << 8) + tid;
    const float* cb = coords + (size_t)b * 3 * NPTS;

    const float px = cb[n];
    const float py = cb[NPTS + n];
    const float pz = cb[2 * NPTS + n];

    float bd[KNN];
    int   bi[KNN];
#pragma unroll
    for (int s = 0; s < KNN; ++s) { bd[s] = 3.4e38f; bi[s] = 0; }

    for (int j0 = 0; j0 < NPTS; j0 += 256) {
        __syncthreads();
        cs0[tid] = cb[j0 + tid];
        cs1[tid] = cb[NPTS + j0 + tid];
        cs2[tid] = cb[2 * NPTS + j0 + tid];
        __syncthreads();
        for (int jj = 0; jj < 256; ++jj) {
            const float dx = px - cs0[jj];
            const float dy = py - cs1[jj];
            const float dz = pz - cs2[jj];
            const float d2 = dx * dx + dy * dy + dz * dz;
            if (d2 < bd[KNN - 1]) {                   // rare after warm-up
                float d = d2; int id = j0 + jj;
#pragma unroll
                for (int s = 0; s < KNN; ++s) {
                    if (d < bd[s]) {
                        float td = bd[s]; int ti = bi[s];
                        bd[s] = d; bi[s] = id;
                        d = td; id = ti;
                    }
                }
            }
        }
    }

    const size_t rowi = ((size_t)b * NPTS + n) * KNN;
#pragma unroll
    for (int k = 0; k < KNN; ++k) {
        const int j = bi[k];
        idx_out[rowi + k] = j;
        lc_out[(((size_t)b * 3 + 0) * NPTS + n) * KNN + k] = px - cb[j];
        lc_out[(((size_t)b * 3 + 1) * NPTS + n) * KNN + k] = py - cb[NPTS + j];
        lc_out[(((size_t)b * 3 + 2) * NPTS + n) * KNN + k] = pz - cb[2 * NPTS + j];
    }
}

// ---------------------------------------------------------------------------
// WMMA fragment loaders (contiguous k order, same in A and B), f16, 16x16x32.
// A (16xK, row-major [m][CK] in LDS): lane<16 -> M=lane,   K = kk*32 + 0..15
//                                     lane>=16 -> M=lane-16, K = kk*32 + 16..31
// B (Kx16, col-major [col][CK] LDS):  lane<16 -> N=lane,   K = kk*32 + 0..15
//                                     lane>=16 -> N=lane-16, K = kk*32 + 16..31
// ---------------------------------------------------------------------------
__device__ __forceinline__ v16h load_a_frag(const _Float16* wl, int CK,
                                            int mrow, int kk, int lane) {
    const int m  = mrow + (lane & 15);
    const int kb = kk * 32 + ((lane & 16) ? 16 : 0);
    const _Float16* p = wl + m * CK + kb;
    v8h lo = *(const v8h*)(p);        // K = kb .. kb+7
    v8h hi = *(const v8h*)(p + 8);    // K = kb+8 .. kb+15
    v16h r;
#pragma unroll
    for (int i = 0; i < 8; ++i) { r[i] = lo[i]; r[8 + i] = hi[i]; }
    return r;
}

__device__ __forceinline__ v16h load_b_frag(const _Float16* buf, int CK,
                                            int coltile, int kk, int lane) {
    const int col = coltile + (lane & 15);
    const int kb  = kk * 32 + ((lane & 16) ? 16 : 0);
    return *(const v16h*)(buf + col * CK + kb);   // 16 contiguous halves (32B)
}

// ---------------------------------------------------------------------------
// Kernel 2: fused gather -> GEMM1 -> BN/ReLU -> GEMM2 -> BN/ReLU ->
//           knn_mlp_x store + max-over-K (y).  One block per (batch, 8 points).
//           Each wave owns one point (16 (n,k) columns).
// ---------------------------------------------------------------------------
__global__ __launch_bounds__(256) void fused_mlp_kernel(
    const float* __restrict__ x,       // [B, CIN, N]
    const int*   __restrict__ idx,     // [B, N, K]
    const float* __restrict__ w1,      // [COUT, CIN]
    const float* __restrict__ g1, const float* __restrict__ b1,
    const float* __restrict__ m1, const float* __restrict__ v1,
    const float* __restrict__ w2,      // [COUT, COUT]
    const float* __restrict__ g2, const float* __restrict__ b2,
    const float* __restrict__ m2, const float* __restrict__ v2,
    float* __restrict__ out_y,         // [B, COUT, N]
    float* __restrict__ out_mlp)       // [B, COUT, N, K]
{
    __shared__ int      idx_s[128];
    __shared__ _Float16 xg  [128 * CIN];    // [col][cin]   16 KB
    __shared__ _Float16 w1l [COUT * CIN];   // [m][cin]     16 KB
    __shared__ _Float16 w2l [COUT * COUT];  // [m][cin2]    32 KB
    __shared__ _Float16 hbuf[128 * COUT];   // [col][ch]    32 KB
    __shared__ float inv1s[COUT], add1s[COUT], inv2s[COUT], add2s[COUT];

    const int tid  = threadIdx.x;
    const int b    = blockIdx.x >> 8;       // 256 tiles per batch
    const int tile = blockIdx.x & 255;
    const int n0   = tile * 8;

    // ---- stage 1: idx, BN affine params, f16 weights --------------------
    if (tid < 128) {
        idx_s[tid] = idx[((size_t)b * NPTS + n0 + (tid >> 4)) * KNN + (tid & 15)];
        const float iv1 = g1[tid] * rsqrtf(v1[tid] + EPS);
        inv1s[tid] = iv1; add1s[tid] = b1[tid] - m1[tid] * iv1;
        const float iv2 = g2[tid] * rsqrtf(v2[tid] + EPS);
        inv2s[tid] = iv2; add2s[tid] = b2[tid] - m2[tid] * iv2;
    }
    for (int e = tid; e < COUT * CIN;  e += 256) w1l[e] = (_Float16)w1[e];
    for (int e = tid; e < COUT * COUT; e += 256) w2l[e] = (_Float16)w2[e];
    __syncthreads();

    // ---- stage 2: gather features into column-major f16 LDS -------------
    for (int e = tid; e < 128 * CIN; e += 256) {
        const int col = e >> 6, ch = e & 63;
        xg[e] = (_Float16)x[((size_t)b * CIN + ch) * NPTS + idx_s[col]];
    }
    __syncthreads();

    const int wv      = tid >> 5;           // wave id = point-in-tile
    const int lane    = tid & 31;
    const int coltile = wv * 16;
    const int off     = (lane & 16) ? 8 : 0;   // C/D M-offset for hi lanes
    const int n       = n0 + wv;
    const int kcol    = lane & 15;          // this lane's K index / D column

    // ---- GEMM1: [COUT x CIN] x [CIN x 16cols], acc f32 ------------------
    v8f acc[8];
#pragma unroll
    for (int r = 0; r < 8; ++r) acc[r] = (v8f)(0.0f);
#pragma unroll
    for (int kk = 0; kk < 2; ++kk) {
        const v16h bf = load_b_frag(xg, CIN, coltile, kk, lane);
#pragma unroll
        for (int r = 0; r < 8; ++r) {
            const v16h af = load_a_frag(w1l, CIN, r * 16, kk, lane);
            acc[r] = __builtin_amdgcn_wmma_f32_16x16x32_f16(
                false, af, false, bf, (short)0, acc[r], false, false);
        }
    }

    // ---- BN1 + ReLU -> hbuf (f16, col-major; wave-private columns) ------
    const int mycol = coltile + kcol;
#pragma unroll
    for (int r = 0; r < 8; ++r) {
        const int ch0 = r * 16 + off;
        const v4f i0 = *(const v4f*)&inv1s[ch0], i1 = *(const v4f*)&inv1s[ch0 + 4];
        const v4f a0 = *(const v4f*)&add1s[ch0], a1 = *(const v4f*)&add1s[ch0 + 4];
        v8h hp;
#pragma unroll
        for (int v = 0; v < 8; ++v) {
            const float inv = (v < 4) ? i0[v] : i1[v - 4];
            const float ad  = (v < 4) ? a0[v] : a1[v - 4];
            float t = acc[r][v] * inv + ad;
            hp[v] = (_Float16)fmaxf(t, 0.0f);
        }
        *(v8h*)&hbuf[mycol * COUT + ch0] = hp;   // ds_store_b128
    }

    // ---- GEMM2: [COUT x COUT] x [COUT x 16cols] -------------------------
    v8f acc2[8];
#pragma unroll
    for (int r = 0; r < 8; ++r) acc2[r] = (v8f)(0.0f);
#pragma unroll
    for (int kk = 0; kk < 4; ++kk) {
        const v16h bf = load_b_frag(hbuf, COUT, coltile, kk, lane);
#pragma unroll
        for (int r = 0; r < 8; ++r) {
            const v16h af = load_a_frag(w2l, COUT, r * 16, kk, lane);
            acc2[r] = __builtin_amdgcn_wmma_f32_16x16x32_f16(
                false, af, false, bf, (short)0, acc2[r], false, false);
        }
    }

    // ---- BN2 + ReLU -> knn_mlp_x store + max-over-K -> y ----------------
#pragma unroll
    for (int r = 0; r < 8; ++r) {
        const int ch0 = r * 16 + off;
        const v4f i0 = *(const v4f*)&inv2s[ch0], i1 = *(const v4f*)&inv2s[ch0 + 4];
        const v4f a0 = *(const v4f*)&add2s[ch0], a1 = *(const v4f*)&add2s[ch0 + 4];
#pragma unroll
        for (int v = 0; v < 8; ++v) {
            const float inv = (v < 4) ? i0[v] : i1[v - 4];
            const float ad  = (v < 4) ? a0[v] : a1[v - 4];
            const float t   = fmaxf(acc2[r][v] * inv + ad, 0.0f);
            const int   ch  = ch0 + v;
            out_mlp[(((size_t)b * COUT + ch) * NPTS + n) * KNN + kcol] = t;
            // max over the 16 K-lanes (shfl stays inside each 16-lane group)
            float mx = t;
            mx = fmaxf(mx, __shfl_xor(mx, 1, 32));
            mx = fmaxf(mx, __shfl_xor(mx, 2, 32));
            mx = fmaxf(mx, __shfl_xor(mx, 4, 32));
            mx = fmaxf(mx, __shfl_xor(mx, 8, 32));
            if (kcol == 0)
                out_y[((size_t)b * COUT + ch) * NPTS + n] = mx;
        }
    }
}

// ---------------------------------------------------------------------------
extern "C" void kernel_launch(void* const* d_in, const int* in_sizes, int n_in,
                              void* d_out, int out_size, void* d_ws, size_t ws_size,
                              hipStream_t stream) {
    const float* x      = (const float*)d_in[0];
    const float* coords = (const float*)d_in[1];
    const float* w1     = (const float*)d_in[2];
    const float* g1     = (const float*)d_in[3];
    const float* b1     = (const float*)d_in[4];
    const float* m1     = (const float*)d_in[5];
    const float* v1     = (const float*)d_in[6];
    const float* w2     = (const float*)d_in[7];
    const float* g2     = (const float*)d_in[8];
    const float* b2     = (const float*)d_in[9];
    const float* m2     = (const float*)d_in[10];
    const float* v2     = (const float*)d_in[11];

    float* out = (float*)d_out;
    float* out_y   = out;                                   // [8,128,2048]
    float* out_mlp = out + (size_t)BATCH * COUT * NPTS;     // [8,128,2048,16]
    float* out_lc  = out_mlp + (size_t)BATCH * COUT * NPTS * KNN; // [8,3,2048,16]

    int* idx_ws = (int*)d_ws;                               // [8,2048,16] = 2 MB? (1 MB)

    knn_local_kernel<<<BATCH * (NPTS / 256), 256, 0, stream>>>(coords, idx_ws, out_lc);

    fused_mlp_kernel<<<BATCH * (NPTS / 8), 256, 0, stream>>>(
        x, idx_ws, w1, g1, b1, m1, v1, w2, g2, b2, m2, v2, out_y, out_mlp);
}
